// SelfAttention_44779329028352
// MI455X (gfx1250) — matrix operand order repaired and checked by reading, not compiled, not run on hardware
//
#include <hip/hip_runtime.h>

// MI455X / gfx1250 flash-attention:
//   scores = x @ x^T (B=8, S=2048, D=512 fp32), mask = padded-key | diagonal,
//   out = softmax(scores) @ x
// Pre-pass converts x -> bf16 (row-major + transposed) in d_ws so the hot loop
// does pure b128 loads + v_wmma_f32_16x16x32_bf16, f32 online softmax.
// DC=128 per wave (DSPLIT=4), amdgpu_waves_per_eu(1) to avoid any spilling.

typedef __attribute__((ext_vector_type(16))) __bf16 v16bf;
typedef __attribute__((ext_vector_type(8)))  float  v8f;
typedef __attribute__((ext_vector_type(4)))  float  f4;
typedef __attribute__((ext_vector_type(4)))  int    i4;
typedef unsigned short ushort_t;
typedef unsigned char  uchar_t;

#define DEV static __device__ __forceinline__

constexpr int Bn = 8, S = 2048, Dm = 512;

DEV ushort_t f2bf(float f) {
    union { float f; unsigned u; } a; a.f = f;
    unsigned r = a.u + 0x7FFFu + ((a.u >> 16) & 1u);   // round-to-nearest-even
    return (ushort_t)(r >> 16);
}

union AB {                       // one 16-bf16 WMMA operand fragment per lane
    v16bf v;
    ushort_t us[16];
    i4 q[2];
};

// ---------------------------------------------------------------------------
// Pre-pass: x(f32) -> xbf [b][s][d] bf16  and  xT [b][d][s] bf16 (tiled transpose)
// ---------------------------------------------------------------------------
__global__ __launch_bounds__(256)
void cvt_kernel(const float* __restrict__ x,
                ushort_t* __restrict__ xbf,
                ushort_t* __restrict__ xT)
{
    __shared__ ushort_t tile[64 * 65];           // [d][s], padded
    const int t  = threadIdx.x;
    const int b  = blockIdx.z;
    const int s0 = blockIdx.y * 64;
    const int d0 = blockIdx.x * 64;

    #pragma unroll 4
    for (int i = 0; i < 16; ++i) {
        const int r = (t >> 6) + i * 4;          // s offset
        const int c = t & 63;                    // d offset
        const float v = x[((size_t)b * S + s0 + r) * Dm + d0 + c];
        const ushort_t h = f2bf(v);
        xbf[((size_t)b * S + s0 + r) * Dm + d0 + c] = h;
        tile[c * 65 + r] = h;
    }
    __syncthreads();
    #pragma unroll 4
    for (int i = 0; i < 16; ++i) {
        const int r = (t >> 6) + i * 4;          // d offset
        const int c = t & 63;                    // s offset
        xT[((size_t)b * Dm + d0 + r) * S + s0 + c] = tile[r * 65 + c];
    }
}

// ---------------------------------------------------------------------------
// Main kernel: 1 wave / block, 16 queries x 128-dim out chunk, keys in 32-chunks
// ---------------------------------------------------------------------------
__global__ __launch_bounds__(32)
__attribute__((amdgpu_waves_per_eu(1)))
void attn_fa_bf16_kernel(const ushort_t* __restrict__ xbf,
                         const ushort_t* __restrict__ xT,
                         const uchar_t* __restrict__ xmask,
                         float* __restrict__ out)
{
    constexpr int KC = 32;            // keys per chunk
    constexpr int DC = 128;           // output D-chunk per wave (DSPLIT = 4)
    constexpr int DT = DC / 16;       // 8 output column tiles -> 64 acc VGPRs
    constexpr int PSTR = 40;
    constexpr float NEG = -1e30f;

    __shared__ alignas(16) ushort_t Pm[16 * PSTR];   // probs, row-major [m][k], bf16

    const int lane = threadIdx.x & 31;
    const int half = lane >> 4;
    const int l16  = lane & 15;

    const int b      = blockIdx.z;
    const int dstart = blockIdx.y * DC;
    const int qbw    = blockIdx.x * 16;

    const ushort_t* xb  = xbf + (size_t)b * S * Dm;
    const ushort_t* xtb = xT  + (size_t)b * Dm * S;
    const uchar_t*  mb  = xmask + (size_t)b * S;

    v8f acc[DT];
    #pragma unroll
    for (int dt = 0; dt < DT; ++dt)
        #pragma unroll
        for (int i = 0; i < 8; ++i) acc[dt][i] = 0.0f;

    float runmax[8], runsum[8];
    #pragma unroll
    for (int r = 0; r < 8; ++r) { runmax[r] = NEG; runsum[r] = 0.0f; }

    const ushort_t* qrow  = xb + (size_t)(qbw + l16) * Dm;          // A rows (matmul1)
    const ushort_t* krow0 = xb + (size_t)l16 * Dm;                  // B rows base (matmul1)
    const ushort_t* vrow0 = xtb + (size_t)(dstart + l16) * S;       // B rows base (matmul2)

    for (int kb = 0; kb < S; kb += KC) {
        // prefetch next key chunk (keys + transposed V rows)
        if (kb + KC < S) {
            __builtin_prefetch(krow0 + (size_t)(kb + KC) * Dm, 0, 3);
            __builtin_prefetch(krow0 + (size_t)(kb + KC + 16) * Dm, 0, 3);
            __builtin_prefetch(vrow0 + kb + KC, 0, 3);
        }

        // ---- matmul 1: 16x32 score tile, K over D=512 ----
        v8f c0, c1;
        #pragma unroll
        for (int i = 0; i < 8; ++i) { c0[i] = 0.0f; c1[i] = 0.0f; }

        const ushort_t* krow = krow0 + (size_t)kb * Dm;
        #pragma unroll 2
        for (int ks = 0; ks < Dm / 32; ++ks) {
            const int dbase = ks * 32;
            AB a;
            const ushort_t* ap = qrow + dbase + half * 8;
            a.q[0] = *(const i4*)ap;
            a.q[1] = *(const i4*)(ap + 16);
            #pragma unroll
            for (int nt = 0; nt < 2; ++nt) {
                const ushort_t* bp = krow + nt * 16 * Dm + dbase + half * 8;
                AB bb;
                bb.q[0] = *(const i4*)bp;
                bb.q[1] = *(const i4*)(bp + 16);
                if (nt == 0)
                    c0 = __builtin_amdgcn_wmma_f32_16x16x32_bf16(false, a.v, false, bb.v,
                                                                 (short)0, c0, false, false);
                else
                    c1 = __builtin_amdgcn_wmma_f32_16x16x32_bf16(false, a.v, false, bb.v,
                                                                 (short)0, c1, false, false);
            }
        }

        // ---- mask + online softmax (half-wave butterfly row reductions) ----
        const int n0 = kb + l16;
        const int n1 = kb + 16 + l16;
        const bool pk0 = mb[n0] != 0;
        const bool pk1 = mb[n1] != 0;

        #pragma unroll
        for (int r = 0; r < 8; ++r) {
            const int m = qbw + r + half * 8;
            float s0 = (pk0 || (n0 == m)) ? NEG : c0[r];
            float s1 = (pk1 || (n1 == m)) ? NEG : c1[r];

            float cm = fmaxf(s0, s1);
            #pragma unroll
            for (int off = 1; off < 16; off <<= 1)
                cm = fmaxf(cm, __shfl_xor(cm, off, 32));

            const float nm = fmaxf(runmax[r], cm);
            const float sc = __expf(runmax[r] - nm);
            runmax[r] = nm;

            const float p0 = __expf(s0 - nm);
            const float p1 = __expf(s1 - nm);
            float ss = p0 + p1;
            #pragma unroll
            for (int off = 1; off < 16; off <<= 1)
                ss += __shfl_xor(ss, off, 32);
            runsum[r] = runsum[r] * sc + ss;

            #pragma unroll
            for (int dt = 0; dt < DT; ++dt) acc[dt][r] *= sc;

            const int row = r + half * 8;
            Pm[row * PSTR + l16]      = f2bf(p0);
            Pm[row * PSTR + 16 + l16] = f2bf(p1);
        }
        __syncthreads();   // single-wave WG: S_NOP in HW, keeps compiler ordering

        // ---- matmul 2: acc(16xDC) += P(16x32) . V(32xDC), V from xT rows ----
        AB a2;
        a2.q[0] = *(const i4*)(Pm + l16 * PSTR + half * 8);
        a2.q[1] = *(const i4*)(Pm + l16 * PSTR + 16 + half * 8);
        const ushort_t* vbase = vrow0 + kb + half * 8;
        #pragma unroll 4
        for (int dt = 0; dt < DT; ++dt) {
            const ushort_t* vp = vbase + (size_t)dt * 16 * S;   // 64KB steps -> inst offsets
            AB b2;
            b2.q[0] = *(const i4*)vp;
            b2.q[1] = *(const i4*)(vp + 16);
            acc[dt] = __builtin_amdgcn_wmma_f32_16x16x32_bf16(false, a2.v, false, b2.v,
                                                              (short)0, acc[dt], false, false);
        }
        __syncthreads();
    }

    // ---- normalize, write fp32 output ----
    #pragma unroll
    for (int dt = 0; dt < DT; ++dt) {
        #pragma unroll
        for (int r = 0; r < 8; ++r) {
            const int m = qbw + r + half * 8;
            out[((size_t)b * S + m) * Dm + dstart + dt * 16 + l16] = acc[dt][r] / runsum[r];
        }
    }
}

// ---------------------------------------------------------------------------
// Fallback (no workspace): self-contained kernel with inline f32->bf16 conversion
// ---------------------------------------------------------------------------
DEV void cvt4(ushort_t* d, f4 s) {
    d[0] = f2bf(s[0]); d[1] = f2bf(s[1]); d[2] = f2bf(s[2]); d[3] = f2bf(s[3]);
}

__global__ __launch_bounds__(128)
void attn_fa_fallback(const float* __restrict__ x,
                      const uchar_t* __restrict__ xmask,
                      float* __restrict__ out)
{
    constexpr int KC = 32, DC = 128, DT = DC / 16, VSTR = 40, PSTR = 40;
    constexpr float NEG = -1e30f;

    __shared__ alignas(16) ushort_t Vt[DC * VSTR];
    __shared__ alignas(16) ushort_t Pm[4 * 16 * PSTR];

    const int tid = threadIdx.x, wv = tid >> 5, lane = tid & 31;
    const int half = lane >> 4, l16 = lane & 15;
    const int b = blockIdx.z, dstart = blockIdx.y * DC;
    const int qbw = blockIdx.x * 64 + wv * 16;

    const float* xb = x + (size_t)b * S * Dm;
    const uchar_t* mb = xmask + (size_t)b * S;
    ushort_t* pw = Pm + wv * 16 * PSTR;

    v8f acc[DT];
    #pragma unroll
    for (int dt = 0; dt < DT; ++dt)
        #pragma unroll
        for (int i = 0; i < 8; ++i) acc[dt][i] = 0.0f;
    float runmax[8], runsum[8];
    #pragma unroll
    for (int r = 0; r < 8; ++r) { runmax[r] = NEG; runsum[r] = 0.0f; }
    const int qrowA = qbw + l16;

    for (int kb = 0; kb < S; kb += KC) {
        __syncthreads();
        #pragma unroll 4
        for (int i = 0; i < KC; ++i)
            Vt[tid * VSTR + i] = f2bf(xb[(size_t)(kb + i) * Dm + dstart + tid]);
        __syncthreads();

        v8f c0, c1;
        #pragma unroll
        for (int i = 0; i < 8; ++i) { c0[i] = 0.0f; c1[i] = 0.0f; }
        #pragma unroll 2
        for (int ks = 0; ks < Dm / 32; ++ks) {
            const int dbase = ks * 32;
            AB a;
            const float* ap = xb + (size_t)qrowA * Dm + dbase + half * 8;
            cvt4(a.us + 0, *(const f4*)ap);        cvt4(a.us + 4, *(const f4*)(ap + 4));
            cvt4(a.us + 8, *(const f4*)(ap + 16)); cvt4(a.us + 12, *(const f4*)(ap + 20));
            #pragma unroll
            for (int nt = 0; nt < 2; ++nt) {
                const float* bp = xb + (size_t)(kb + nt * 16 + l16) * Dm + dbase + half * 8;
                AB bb;
                cvt4(bb.us + 0, *(const f4*)bp);       cvt4(bb.us + 4, *(const f4*)(bp + 4));
                cvt4(bb.us + 8, *(const f4*)(bp + 16)); cvt4(bb.us + 12, *(const f4*)(bp + 20));
                if (nt == 0)
                    c0 = __builtin_amdgcn_wmma_f32_16x16x32_bf16(false, a.v, false, bb.v,
                                                                 (short)0, c0, false, false);
                else
                    c1 = __builtin_amdgcn_wmma_f32_16x16x32_bf16(false, a.v, false, bb.v,
                                                                 (short)0, c1, false, false);
            }
        }

        const int n0 = kb + l16, n1 = kb + 16 + l16;
        const bool pk0 = mb[n0] != 0, pk1 = mb[n1] != 0;
        #pragma unroll
        for (int r = 0; r < 8; ++r) {
            const int m = qbw + r + half * 8;
            float s0 = (pk0 || (n0 == m)) ? NEG : c0[r];
            float s1 = (pk1 || (n1 == m)) ? NEG : c1[r];
            float cm = fmaxf(s0, s1);
            #pragma unroll
            for (int off = 1; off < 16; off <<= 1)
                cm = fmaxf(cm, __shfl_xor(cm, off, 32));
            const float nm = fmaxf(runmax[r], cm);
            const float sc = __expf(runmax[r] - nm);
            runmax[r] = nm;
            const float p0 = __expf(s0 - nm), p1 = __expf(s1 - nm);
            float ss = p0 + p1;
            #pragma unroll
            for (int off = 1; off < 16; off <<= 1)
                ss += __shfl_xor(ss, off, 32);
            runsum[r] = runsum[r] * sc + ss;
            #pragma unroll
            for (int dt = 0; dt < DT; ++dt) acc[dt][r] *= sc;
            const int row = r + half * 8;
            pw[row * PSTR + l16]      = f2bf(p0);
            pw[row * PSTR + 16 + l16] = f2bf(p1);
        }
        __syncthreads();

        AB a2;
        a2.q[0] = *(const i4*)(pw + l16 * PSTR + half * 8);
        a2.q[1] = *(const i4*)(pw + l16 * PSTR + 16 + half * 8);
        #pragma unroll
        for (int dt = 0; dt < DT; ++dt) {
            const ushort_t* vr = Vt + (dt * 16 + l16) * VSTR + half * 8;
            AB b2;
            b2.q[0] = *(const i4*)vr;
            b2.q[1] = *(const i4*)(vr + 16);
            acc[dt] = __builtin_amdgcn_wmma_f32_16x16x32_bf16(false, a2.v, false, b2.v,
                                                              (short)0, acc[dt], false, false);
        }
    }

    #pragma unroll
    for (int dt = 0; dt < DT; ++dt)
        #pragma unroll
        for (int r = 0; r < 8; ++r) {
            const int m = qbw + r + half * 8;
            out[((size_t)b * S + m) * Dm + dstart + dt * 16 + l16] = acc[dt][r] / runsum[r];
        }
}

// ---------------------------------------------------------------------------
extern "C" void kernel_launch(void* const* d_in, const int* in_sizes, int n_in,
                              void* d_out, int out_size, void* d_ws, size_t ws_size,
                              hipStream_t stream) {
    (void)in_sizes; (void)n_in; (void)out_size;
    const float* x = (const float*)d_in[0];
    const uchar_t* xm = (const uchar_t*)d_in[1];
    float* out = (float*)d_out;

    const size_t nElem = (size_t)Bn * S * Dm;
    const size_t wsNeed = 2 * nElem * sizeof(ushort_t);   // xbf + xT, 32 MiB

    if (d_ws != nullptr && ws_size >= wsNeed) {
        ushort_t* xbf = (ushort_t*)d_ws;
        ushort_t* xT  = xbf + nElem;
        cvt_kernel<<<dim3(Dm / 64, S / 64, Bn), dim3(256), 0, stream>>>(x, xbf, xT);
        attn_fa_bf16_kernel<<<dim3(S / 16, 4, Bn), dim3(32), 0, stream>>>(xbf, xT, xm, out);
    } else {
        attn_fa_fallback<<<dim3(S / 64, Dm / 128, Bn), dim3(128), 0, stream>>>(x, xm, out);
    }
}
